// LinearAttention_59485297050197
// MI455X (gfx1250) — compile-verified, matrix operand order checked by reading, not run on hardware
//
#include <hip/hip_runtime.h>
#include <hip/hip_bf16.h>
#include <math.h>

typedef __bf16 bf16;
typedef __attribute__((ext_vector_type(16))) __bf16 v16bf;
typedef __attribute__((ext_vector_type(8)))  __bf16 v8bf;
typedef __attribute__((ext_vector_type(4)))  __bf16 v4bf;
typedef __attribute__((ext_vector_type(8)))  float   v8f;

constexpr int B_ = 8, H_ = 16, S_ = 4096, D_ = 64;
constexpr int TS    = 128;         // s-tile rows
constexpr int SPLIT = 4;           // S split for phase 1 (partials stay L2-resident)
constexpr int SB    = S_ / SPLIT;  // 1024 s per phase-1 block
constexpr int LSTR  = TS + 8;      // 136 bf16 row stride = 272 B (16B aligned rows)
constexpr float ANG  = 3.14159265358979323846f * 0.5f / (float)S_;
constexpr float TINV = 0.125f;     // 1 / TEMPERATURE

__device__ __forceinline__ float nonneg(float x) {
    return x < 0.0f ? __expf(x) : (x + 1.0f);
}

// Build a 16x32 bf16 WMMA fragment from an LDS row (K contiguous per lane).
// lanes 0-15: elems 0..7 <- K[k0..k0+7],  elems 8..15 <- K[k0+16..k0+23]
// lanes 16-31 pass k0+8 so the halves land on K 8..15 / 24..31 per the ISA layout.
__device__ __forceinline__ v16bf frag(const bf16* row, int k0) {
    v8bf lo = *(const v8bf*)(row + k0);
    v8bf hi = *(const v8bf*)(row + k0 + 16);
    v16bf r;
#pragma unroll
    for (int j = 0; j < 8; ++j) { r[j] = lo[j]; r[8 + j] = hi[j]; }
    return r;
}

// ---------------------------------------------------------------------------
// Phase 1: KV_cos / KV_sin partials.  grid = (B*H, SPLIT), block = 256 (8 waves)
// kvp layout: [SPLIT][B*H][2][64][64] f32
// ---------------------------------------------------------------------------
__global__ __launch_bounds__(256) void lin_attn_kv(const float* __restrict__ k,
                                                   const float* __restrict__ v,
                                                   float* __restrict__ kvp) {
    __shared__ bf16 kT [D_][LSTR];   // kf  transposed: [d][s]
    __shared__ bf16 vcT[D_][LSTR];   // v*cos transposed: [e][s]
    __shared__ bf16 vsT[D_][LSTR];   // v*sin transposed: [e][s]

    const int bh    = blockIdx.x;
    const int split = blockIdx.y;
    const int tid   = threadIdx.x;
    const int wave  = tid >> 5;
    const int lane  = tid & 31;
    const int l16   = lane & 15;
    const int hi    = lane >> 4;
    const int mt      = wave & 3;    // d-tile (rows of KV)
    const int use_sin = wave >> 2;   // waves 0-3: cos, 4-7: sin

    const float* kb = k + (size_t)bh * S_ * D_;
    const float* vb = v + (size_t)bh * S_ * D_;

    v8f acc0 = {}, acc1 = {}, acc2 = {}, acc3 = {};

    for (int t = 0; t < SB / TS; ++t) {
        const int s0 = split * SB + t * TS;

        if (t + 1 < SB / TS) {  // prefetch next tile (global_prefetch_b8)
            const size_t nxt = (size_t)(s0 + TS) * D_ + (size_t)tid * 32;
            __builtin_prefetch(kb + nxt, 0, 1);
            __builtin_prefetch(vb + nxt, 0, 1);
        }

        // Stage feature-mapped tiles into LDS, transposed, bf16.
#pragma unroll
        for (int i = 0; i < 8; ++i) {
            const int idx = i * 256 + tid;       // float4 index over 128s x 16(d/4)
            const int d4  = idx & 15;
            const int s   = idx >> 4;
            const float4 k4 = *(const float4*)(kb + (size_t)(s0 + s) * D_ + d4 * 4);
            const float4 v4 = *(const float4*)(vb + (size_t)(s0 + s) * D_ + d4 * 4);
            float sn, cs;
            __sincosf(ANG * (float)(s0 + s), &sn, &cs);
            const int d = d4 * 4;
            kT [d + 0][s] = (bf16)nonneg(k4.x);
            kT [d + 1][s] = (bf16)nonneg(k4.y);
            kT [d + 2][s] = (bf16)nonneg(k4.z);
            kT [d + 3][s] = (bf16)nonneg(k4.w);
            vcT[d + 0][s] = (bf16)(v4.x * cs);
            vcT[d + 1][s] = (bf16)(v4.y * cs);
            vcT[d + 2][s] = (bf16)(v4.z * cs);
            vcT[d + 3][s] = (bf16)(v4.w * cs);
            vsT[d + 0][s] = (bf16)(v4.x * sn);
            vsT[d + 1][s] = (bf16)(v4.y * sn);
            vsT[d + 2][s] = (bf16)(v4.z * sn);
            vsT[d + 3][s] = (bf16)(v4.w * sn);
        }
        __syncthreads();

        bf16 (*Bt)[LSTR] = use_sin ? vsT : vcT;
#pragma unroll
        for (int kc = 0; kc < 4; ++kc) {          // 4 x K=32 chunks of the s-tile
            const int k0 = kc * 32 + hi * 8;
            const v16bf a  = frag(&kT[mt * 16 + l16][0], k0);
            const v16bf b0 = frag(&Bt[ 0 + l16][0], k0);
            const v16bf b1 = frag(&Bt[16 + l16][0], k0);
            const v16bf b2 = frag(&Bt[32 + l16][0], k0);
            const v16bf b3 = frag(&Bt[48 + l16][0], k0);
            acc0 = __builtin_amdgcn_wmma_f32_16x16x32_bf16(false, a, false, b0, (short)0, acc0, false, false);
            acc1 = __builtin_amdgcn_wmma_f32_16x16x32_bf16(false, a, false, b1, (short)0, acc1, false, false);
            acc2 = __builtin_amdgcn_wmma_f32_16x16x32_bf16(false, a, false, b2, (short)0, acc2, false, false);
            acc3 = __builtin_amdgcn_wmma_f32_16x16x32_bf16(false, a, false, b3, (short)0, acc3, false, false);
        }
        __syncthreads();
    }

    float* outp = kvp + (((size_t)split * (B_ * H_) + bh) * 2 + use_sin) * (D_ * D_);
#pragma unroll
    for (int i = 0; i < 8; ++i) {
        const int row = mt * 16 + hi * 8 + i;     // C/D layout: lanes16-31 -> M+8
        outp[row * D_ +  0 + l16] = acc0[i];
        outp[row * D_ + 16 + l16] = acc1[i];
        outp[row * D_ + 32 + l16] = acc2[i];
        outp[row * D_ + 48 + l16] = acc3[i];
    }
}

// ---------------------------------------------------------------------------
// Phase 2: out = [q_cos|q_sin] @ [KV_cos;KV_sin] / 8.
// grid = (B*H, S/TS), block = 256 (8 waves, one 16-row m-tile each)
// ---------------------------------------------------------------------------
__global__ __launch_bounds__(256) void lin_attn_out(const float* __restrict__ q,
                                                    const unsigned char* __restrict__ tmask,
                                                    const float* __restrict__ kvp,
                                                    float* __restrict__ out) {
    __shared__ bf16 qA [TS][LSTR];   // [s][d'], d' in [0,128): cos | sin
    __shared__ bf16 kvT[D_][LSTR];   // [e][d'] (B transposed: column per lane)

    const int bh  = blockIdx.x;
    const int b   = bh / H_;
    const int s0  = blockIdx.y * TS;
    const int tid = threadIdx.x;
    const int wave = tid >> 5;
    const int lane = tid & 31;
    const int l16  = lane & 15;
    const int hi   = lane >> 4;

    // Stage combined KV = sum of SPLIT partials (L2-resident), bf16, transposed.
#pragma unroll 4
    for (int i = 0; i < 32; ++i) {
        const int idx = i * 256 + tid;            // over 128(k') x 64(e)
        const int e  = idx & 63;
        const int kp = idx >> 6;                  // 0..63 cos rows, 64..127 sin rows
        float sum = 0.0f;
#pragma unroll
        for (int sp = 0; sp < SPLIT; ++sp)
            sum += kvp[((size_t)sp * (B_ * H_) + bh) * (2 * D_ * D_) + (size_t)kp * D_ + e];
        kvT[e][kp] = (bf16)sum;
    }

    // Stage masked q features: row-major [s][d'] (K contiguous per lane).
    const float* qb = q + (size_t)bh * S_ * D_;
#pragma unroll
    for (int i = 0; i < 8; ++i) {
        const int idx = i * 256 + tid;
        const int d4  = idx & 15;
        const int s   = idx >> 4;
        const float4 q4 = *(const float4*)(qb + (size_t)(s0 + s) * D_ + d4 * 4);
        float sn, cs;
        __sincosf(ANG * (float)(s0 + s), &sn, &cs);
        const float m  = tmask[(size_t)b * S_ + s0 + s] ? 1.0f : 0.0f;
        const float f0 = nonneg(q4.x) * m, f1 = nonneg(q4.y) * m;
        const float f2 = nonneg(q4.z) * m, f3 = nonneg(q4.w) * m;
        const int d = d4 * 4;
        v4bf qc = { (bf16)(f0 * cs), (bf16)(f1 * cs), (bf16)(f2 * cs), (bf16)(f3 * cs) };
        v4bf qs = { (bf16)(f0 * sn), (bf16)(f1 * sn), (bf16)(f2 * sn), (bf16)(f3 * sn) };
        *(v4bf*)&qA[s][d]       = qc;
        *(v4bf*)&qA[s][64 + d]  = qs;
    }
    __syncthreads();

    v8f acc0 = {}, acc1 = {}, acc2 = {}, acc3 = {};
#pragma unroll
    for (int kc = 0; kc < 4; ++kc) {              // K = 128 over 4 chunks
        const int k0 = kc * 32 + hi * 8;
        const v16bf a  = frag(&qA[wave * 16 + l16][0], k0);
        const v16bf b0 = frag(&kvT[ 0 + l16][0], k0);
        const v16bf b1 = frag(&kvT[16 + l16][0], k0);
        const v16bf b2 = frag(&kvT[32 + l16][0], k0);
        const v16bf b3 = frag(&kvT[48 + l16][0], k0);
        acc0 = __builtin_amdgcn_wmma_f32_16x16x32_bf16(false, a, false, b0, (short)0, acc0, false, false);
        acc1 = __builtin_amdgcn_wmma_f32_16x16x32_bf16(false, a, false, b1, (short)0, acc1, false, false);
        acc2 = __builtin_amdgcn_wmma_f32_16x16x32_bf16(false, a, false, b2, (short)0, acc2, false, false);
        acc3 = __builtin_amdgcn_wmma_f32_16x16x32_bf16(false, a, false, b3, (short)0, acc3, false, false);
    }

    float* ob = out + (size_t)bh * S_ * D_;
#pragma unroll
    for (int i = 0; i < 8; ++i) {
        const int row = s0 + wave * 16 + hi * 8 + i;
        ob[(size_t)row * D_ +  0 + l16] = acc0[i] * TINV;
        ob[(size_t)row * D_ + 16 + l16] = acc1[i] * TINV;
        ob[(size_t)row * D_ + 32 + l16] = acc2[i] * TINV;
        ob[(size_t)row * D_ + 48 + l16] = acc3[i] * TINV;
    }
}

extern "C" void kernel_launch(void* const* d_in, const int* in_sizes, int n_in,
                              void* d_out, int out_size, void* d_ws, size_t ws_size,
                              hipStream_t stream) {
    (void)in_sizes; (void)n_in; (void)out_size; (void)ws_size;
    const float*         q  = (const float*)d_in[0];
    const float*         k  = (const float*)d_in[1];
    const float*         v  = (const float*)d_in[2];
    const unsigned char* tm = (const unsigned char*)d_in[3];  // bool mask (1 byte)
    float* out = (float*)d_out;
    float* kvp = (float*)d_ws;  // SPLIT*B*H*2*64*64 f32 = 16 MB scratch

    lin_attn_kv <<<dim3(B_ * H_, SPLIT),   256, 0, stream>>>(k, v, kvp);
    lin_attn_out<<<dim3(B_ * H_, S_ / TS), 256, 0, stream>>>(q, tm, kvp, out);
}
